// NoPEAttention_26164940767697
// MI455X (gfx1250) — compile-verified, matrix operand order checked by reading, not run on hardware
//
#include <hip/hip_runtime.h>
#include <stdint.h>

#define B_   2
#define S_   2048
#define H_   2048
#define NH_  16
#define HD_  128
#define MROWS (B_ * S_)   // 4096
#define NQKV  (3 * H_)    // 6144

typedef __bf16 bf16;
typedef __attribute__((ext_vector_type(16))) __bf16 v16bf;
typedef __attribute__((ext_vector_type(8)))  __bf16 v8bf;
typedef __attribute__((ext_vector_type(8)))  float  v8f;
typedef __attribute__((ext_vector_type(4)))  int    v4i;

union FragBF { v16bf v; v8bf h[2]; };
union PackBF { v8bf v; bf16 e[8]; };

#if defined(__has_builtin)
#  if __has_builtin(__builtin_amdgcn_global_load_async_to_lds_b128)
#    define HAVE_ASYNC_LDS 1
#  endif
#  if __has_builtin(__builtin_amdgcn_s_wait_asynccnt)
#    define HAVE_WAIT_ASYNC 1
#  endif
#endif
#ifndef HAVE_ASYNC_LDS
#  define HAVE_ASYNC_LDS 0
#endif
#ifndef HAVE_WAIT_ASYNC
#  define HAVE_WAIT_ASYNC 0
#endif

static __device__ __forceinline__ bf16 f2bf(float f) {
  uint32_t u = __builtin_bit_cast(uint32_t, f);
  uint32_t r = u + 0x7FFFu + ((u >> 16) & 1u);   // round-to-nearest-even
  uint16_t s = (uint16_t)(r >> 16);
  return __builtin_bit_cast(bf16, s);
}

// 16-byte global -> LDS copy; async DMA path when available (ASYNCcnt-tracked).
static __device__ __forceinline__ void copy16B_g2l(const bf16* g, bf16* l) {
#if HAVE_ASYNC_LDS
  __builtin_amdgcn_global_load_async_to_lds_b128((v4i*)g, (v4i*)l, 0, 0);
#else
  *(v8bf*)l = *(const v8bf*)g;
#endif
}

static __device__ __forceinline__ void wait_async_lds() {
#if HAVE_ASYNC_LDS
#  if HAVE_WAIT_ASYNC
  __builtin_amdgcn_s_wait_asynccnt(0);
#  else
  asm volatile("s_wait_asynccnt 0" ::: "memory");
#  endif
#endif
}

// Load a 16x32 bf16 operand fragment (A-layout) from a k-contiguous row.
// lane<16: k = {0..7, 16..23}; lane>=16: k = {8..15, 24..31}  (half8 = lane>>4)
static __device__ __forceinline__ v16bf load_frag(const bf16* rowptr, int half8) {
  const v8bf* p = (const v8bf*)rowptr;
  FragBF u;
  u.h[0] = p[half8];
  u.h[1] = p[2 + half8];
  return u.v;
}

static __device__ __forceinline__ v8f wmma_bf16(v16bf a, v16bf b, v8f c) {
  return __builtin_amdgcn_wmma_f32_16x16x32_bf16(false, a, false, b, (short)0, c,
                                                 false, false);
}

// ---------------------------------------------------------------------------
// flat fp32 -> bf16 (coalesced, 8 elements/thread)
// ---------------------------------------------------------------------------
__global__ __launch_bounds__(256) void convert_f32_bf16(
    const float* __restrict__ in, bf16* __restrict__ out) {
  size_t i = ((size_t)blockIdx.x * 256 + threadIdx.x) * 8;
  float4 a = *(const float4*)(in + i);
  float4 b = *(const float4*)(in + i + 4);
  PackBF p;
  p.e[0] = f2bf(a.x); p.e[1] = f2bf(a.y); p.e[2] = f2bf(a.z); p.e[3] = f2bf(a.w);
  p.e[4] = f2bf(b.x); p.e[5] = f2bf(b.y); p.e[6] = f2bf(b.z); p.e[7] = f2bf(b.w);
  *(v8bf*)(out + i) = p.v;
}

// ---------------------------------------------------------------------------
// fp32 [R][C]  ->  bf16 [C][R]   (tiled, coalesced both sides)
// ---------------------------------------------------------------------------
__global__ __launch_bounds__(256) void transpose_f32_to_bf16(
    const float* __restrict__ in, bf16* __restrict__ out, int R, int C) {
  __shared__ float tile[32][33];
  int c0 = blockIdx.x * 32, r0 = blockIdx.y * 32;
  int x = threadIdx.x, y = threadIdx.y;            // 32 x 8
#pragma unroll
  for (int i = 0; i < 32; i += 8)
    tile[y + i][x] = in[(size_t)(r0 + y + i) * C + (c0 + x)];
  __syncthreads();
#pragma unroll
  for (int i = 0; i < 32; i += 8)
    out[(size_t)(c0 + y + i) * R + (r0 + x)] = f2bf(tile[x][y + i]);
}

// ---------------------------------------------------------------------------
// Unified bf16 WMMA GEMM: C = A[M,2048] * BT[N,2048]^T (+bias)
// Block tile 128x128, BK=32, double-buffered async global->LDS staging.
// EPI 0: QKV epilogue (scatter Q scaled / K as [b,h,s,d], V^T as [b,h,d,s])
// EPI 1: dense epilogue (fp32 out + bias)
// ---------------------------------------------------------------------------
template <int EPI>
__global__ __launch_bounds__(256) void gemm_bf16(
    const bf16* __restrict__ A, const bf16* __restrict__ BT,
    const float* __restrict__ bias,
    bf16* __restrict__ Qo, bf16* __restrict__ Ko, bf16* __restrict__ VTo,
    float* __restrict__ out) {
  const int K = H_;
  const int BUF = 128 * 32;
  __shared__ __align__(16) bf16 As[2 * 128 * 32];
  __shared__ __align__(16) bf16 Bs[2 * 128 * 32];
  int tid = threadIdx.x;
  int m0 = blockIdx.y * 128, n0 = blockIdx.x * 128;
  int lane = tid & 31, w = tid >> 5;
  int wr = w >> 1, wc = w & 1;
  int col = lane & 15, half8 = lane >> 4;
  int srow = tid >> 1, sseg = (tid & 1) * 16;

  const bf16* gA = A + (size_t)(m0 + srow) * K + sseg;
  const bf16* gB = BT + (size_t)(n0 + srow) * K + sseg;
  bf16* lA = &As[srow * 32 + sseg];
  bf16* lB = &Bs[srow * 32 + sseg];

  v8f zero = {};
  v8f acc[2][4];
#pragma unroll
  for (int i = 0; i < 2; ++i)
#pragma unroll
    for (int j = 0; j < 4; ++j) acc[i][j] = zero;

  // prologue: stage k-slab 0 into buffer 0
  copy16B_g2l(gA, lA);      copy16B_g2l(gA + 8, lA + 8);
  copy16B_g2l(gB, lB);      copy16B_g2l(gB + 8, lB + 8);

  for (int kk = 0; kk < K; kk += 32) {
    int cur = (kk >> 5) & 1;
    wait_async_lds();
    __syncthreads();                       // buffer `cur` ready for all waves
    if (kk + 32 < K) {                     // prefetch next slab into cur^1
      int nxt = cur ^ 1;
      const bf16* ga = gA + kk + 32;
      const bf16* gb = gB + kk + 32;
      bf16* la = lA + nxt * BUF;
      bf16* lb = lB + nxt * BUF;
      copy16B_g2l(ga, la);  copy16B_g2l(ga + 8, la + 8);
      copy16B_g2l(gb, lb);  copy16B_g2l(gb + 8, lb + 8);
    }
    const bf16* Ab = &As[cur * BUF];
    const bf16* Bb = &Bs[cur * BUF];
    v16bf af[2], bfr[4];
    af[0] = load_frag(Ab + (wr * 32 + col) * 32, half8);
    af[1] = load_frag(Ab + (wr * 32 + 16 + col) * 32, half8);
#pragma unroll
    for (int ct = 0; ct < 4; ++ct)
      bfr[ct] = load_frag(Bb + (wc * 64 + ct * 16 + col) * 32, half8);
#pragma unroll
    for (int rt = 0; rt < 2; ++rt)
#pragma unroll
      for (int ct = 0; ct < 4; ++ct)
        acc[rt][ct] = wmma_bf16(af[rt], bfr[ct], acc[rt][ct]);
  }

  if (EPI == 0) {
    const float qscale = 0.08838834764831845f;  // 1/sqrt(HD)
#pragma unroll
    for (int ct = 0; ct < 4; ++ct) {
      int n = n0 + wc * 64 + ct * 16 + col;
      float bv = bias[n];
      int which = n >> 11;
      int h = (n & 2047) >> 7;
      int d = n & 127;
#pragma unroll
      for (int rt = 0; rt < 2; ++rt)
#pragma unroll
        for (int r = 0; r < 8; ++r) {
          int m = m0 + wr * 32 + rt * 16 + r + half8 * 8;
          int bb = m >> 11, s = m & 2047;
          float v = acc[rt][ct][r] + bv;
          size_t bh = (size_t)(bb * NH_ + h);
          if (which == 0)      Qo[(bh * S_ + s) * HD_ + d] = f2bf(v * qscale);
          else if (which == 1) Ko[(bh * S_ + s) * HD_ + d] = f2bf(v);
          else                 VTo[(bh * HD_ + d) * (size_t)S_ + s] = f2bf(v);
        }
    }
  } else {
#pragma unroll
    for (int ct = 0; ct < 4; ++ct) {
      int n = n0 + wc * 64 + ct * 16 + col;
      float bv = bias[n];
#pragma unroll
      for (int rt = 0; rt < 2; ++rt)
#pragma unroll
        for (int r = 0; r < 8; ++r) {
          int m = m0 + wr * 32 + rt * 16 + r + half8 * 8;
          out[(size_t)m * H_ + n] = acc[rt][ct][r] + bv;
        }
    }
  }
}

// ---------------------------------------------------------------------------
// Fused causal flash attention: one wave = 16 query rows of one (b,h).
// ---------------------------------------------------------------------------
__global__ __launch_bounds__(32) void attention_fused(
    const bf16* __restrict__ Q, const bf16* __restrict__ Kb,
    const bf16* __restrict__ VT, bf16* __restrict__ Ao) {
  __shared__ __align__(16) bf16 P_lds[16 * 32];
  int lane = threadIdx.x;
  int col = lane & 15, half8 = lane >> 4;
  int qt = blockIdx.x, bh = blockIdx.y;
  int qbase = qt * 16;
  const bf16* Qp = Q  + (size_t)bh * S_ * HD_;
  const bf16* Kp = Kb + (size_t)bh * S_ * HD_;
  const bf16* Vp = VT + (size_t)bh * HD_ * S_;

  v16bf qf[4];
#pragma unroll
  for (int kc = 0; kc < 4; ++kc)
    qf[kc] = load_frag(Qp + (size_t)(qbase + col) * HD_ + kc * 32, half8);

  v8f zero = {};
  v8f o[8];
  float mstat[8], lstat[8];
#pragma unroll
  for (int i = 0; i < 8; ++i) { o[i] = zero; mstat[i] = -__builtin_inff(); lstat[i] = 0.f; }

  int nkeys = qbase + 16;              // causal: keys [0, qbase+15]
  int njb = (nkeys + 31) >> 5;
  for (int jb = 0; jb < njb; ++jb) {
    int k0 = jb * 32;
    // preload all 8 K fragments (2 key half-tiles x 4 k-chunks) so the
    // scheduler can clause the 16 b128 loads and stagger loadcnt waits
    v16bf kf[8];
#pragma unroll
    for (int kc = 0; kc < 4; ++kc) {
      kf[kc]     = load_frag(Kp + (size_t)(k0 + col) * HD_ + kc * 32, half8);
      kf[4 + kc] = load_frag(Kp + (size_t)(k0 + 16 + col) * HD_ + kc * 32, half8);
    }
    v8f s0 = zero, s1 = zero;
#pragma unroll
    for (int kc = 0; kc < 4; ++kc) {
      s0 = wmma_bf16(qf[kc], kf[kc], s0);
      s1 = wmma_bf16(qf[kc], kf[4 + kc], s1);
    }
    // online softmax update (row stats replicated across each 16-lane half)
    v8f av;
#pragma unroll
    for (int r = 0; r < 8; ++r) {
      int qrow = qbase + r + half8 * 8;
      float a = (k0 + col      <= qrow) ? s0[r] : -__builtin_inff();
      float c = (k0 + 16 + col <= qrow) ? s1[r] : -__builtin_inff();
      float t = fmaxf(a, c);
      t = fmaxf(t, __shfl_xor(t, 1));
      t = fmaxf(t, __shfl_xor(t, 2));
      t = fmaxf(t, __shfl_xor(t, 4));
      t = fmaxf(t, __shfl_xor(t, 8));
      float mn = fmaxf(mstat[r], t);
      float alpha = __expf(mstat[r] - mn);
      float e0 = __expf(a - mn);
      float e1 = __expf(c - mn);
      float rs = e0 + e1;
      rs += __shfl_xor(rs, 1);
      rs += __shfl_xor(rs, 2);
      rs += __shfl_xor(rs, 4);
      rs += __shfl_xor(rs, 8);
      lstat[r] = lstat[r] * alpha + rs;
      mstat[r] = mn;
      av[r] = alpha;
      int prow = r + half8 * 8;
      P_lds[prow * 32 + col]      = f2bf(e0);
      P_lds[prow * 32 + 16 + col] = f2bf(e1);
    }
#pragma unroll
    for (int nt = 0; nt < 8; ++nt) o[nt] *= av;   // rescale running output
    __syncthreads();
    v16bf pf = load_frag(&P_lds[col * 32], half8);
    __syncthreads();
    // preload V fragments in two banks of 4 to bound register pressure
#pragma unroll
    for (int g = 0; g < 2; ++g) {
      v16bf vf[4];
#pragma unroll
      for (int j = 0; j < 4; ++j)
        vf[j] = load_frag(Vp + (size_t)((g * 4 + j) * 16 + col) * S_ + k0, half8);
#pragma unroll
      for (int j = 0; j < 4; ++j)
        o[g * 4 + j] = wmma_bf16(pf, vf[j], o[g * 4 + j]);
    }
  }

  int bb = bh / NH_, hh = bh % NH_;
#pragma unroll
  for (int r = 0; r < 8; ++r) {
    int qrow = qbase + r + half8 * 8;
    float inv = 1.0f / lstat[r];
    bf16* dst = Ao + ((size_t)(bb * S_ + qrow)) * H_ + hh * HD_;
#pragma unroll
    for (int nt = 0; nt < 8; ++nt)
      dst[nt * 16 + col] = f2bf(o[nt][r] * inv);
  }
}

// ---------------------------------------------------------------------------
extern "C" void kernel_launch(void* const* d_in, const int* in_sizes, int n_in,
                              void* d_out, int out_size, void* d_ws, size_t ws_size,
                              hipStream_t stream) {
  const float* hidden = (const float*)d_in[0];   // [2,2048,2048]
  const float* Wqkv   = (const float*)d_in[1];   // [2048,6144]
  const float* bqkv   = (const float*)d_in[2];   // [6144]
  const float* Wd     = (const float*)d_in[3];   // [2048,2048]
  const float* bd     = (const float*)d_in[4];   // [2048]
  float* out = (float*)d_out;

  char* ws = (char*)d_ws;
  size_t off = 0;
  bf16* Hb    = (bf16*)(ws + off); off += (size_t)MROWS * H_ * 2;          // 16 MiB
  bf16* WqkvT = (bf16*)(ws + off); off += (size_t)NQKV * H_ * 2;           // 24 MiB
  bf16* WdT   = (bf16*)(ws + off); off += (size_t)H_ * H_ * 2;             //  8 MiB
  bf16* Qw    = (bf16*)(ws + off); off += (size_t)B_ * NH_ * S_ * HD_ * 2; // 16 MiB
  bf16* Kw    = (bf16*)(ws + off); off += (size_t)B_ * NH_ * S_ * HD_ * 2; // 16 MiB
  bf16* VTw   = (bf16*)(ws + off); off += (size_t)B_ * NH_ * S_ * HD_ * 2; // 16 MiB
  bf16* Aow   = (bf16*)(ws + off); off += (size_t)MROWS * H_ * 2;          // 16 MiB

  convert_f32_bf16<<<(MROWS * H_) / (256 * 8), 256, 0, stream>>>(hidden, Hb);
  transpose_f32_to_bf16<<<dim3(NQKV / 32, H_ / 32), dim3(32, 8), 0, stream>>>(
      Wqkv, WqkvT, H_, NQKV);
  transpose_f32_to_bf16<<<dim3(H_ / 32, H_ / 32), dim3(32, 8), 0, stream>>>(
      Wd, WdT, H_, H_);
  gemm_bf16<0><<<dim3(NQKV / 128, MROWS / 128), 256, 0, stream>>>(
      Hb, WqkvT, bqkv, Qw, Kw, VTw, nullptr);
  attention_fused<<<dim3(S_ / 16, B_ * NH_), 32, 0, stream>>>(Qw, Kw, VTw, Aow);
  gemm_bf16<1><<<dim3(H_ / 128, MROWS / 128), 256, 0, stream>>>(
      Aow, WdT, bd, nullptr, nullptr, nullptr, out);
}
